// UniformStrideAttention_48361331752990
// MI455X (gfx1250) — matrix operand order repaired and checked by reading, not compiled, not run on hardware
//
#include <hip/hip_runtime.h>
#include <hip/hip_bf16.h>
#include <math.h>

typedef _Float16 half8  __attribute__((ext_vector_type(8)));
typedef _Float16 half16 __attribute__((ext_vector_type(16)));
typedef float    float8 __attribute__((ext_vector_type(8)));

#define B_   2
#define S_   2048
#define D_   768
#define H_   12
#define HD_  64
#define M_   (B_ * S_)   // 4096 rows
#define N1_  (3 * D_)    // 2304 qkv features

// D = A(16x32 f16) * B(32x16 f16) + C(16x16 f32)
__device__ __forceinline__ float8 wmma16(half16 a, half16 b, float8 c) {
  return __builtin_amdgcn_wmma_f32_16x16x32_f16(false, a, false, b, (short)0, c,
                                                false, false);
}

__device__ __forceinline__ half16 pack2(const _Float16* lo_p, const _Float16* hi_p) {
  half8 lo = *(const half8*)lo_p;
  half8 hi = *(const half8*)hi_p;
  half16 r;
#pragma unroll
  for (int i = 0; i < 8; ++i) { r[i] = lo[i]; r[i + 8] = hi[i]; }
  return r;
}

// A-matrix 16x32 fragment from row-major [*, ld] f16 source.
// Layout: lanes 0-15 row m=lane, K 0..15; lanes 16-31 K 16..31.
__device__ __forceinline__ half16 load_afrag(const _Float16* base, int ld, int lane) {
  const int m  = lane & 15;
  const int ko = (lane & 16) ? 16 : 0;
  const _Float16* p = base + (size_t)m * ld + ko;
  return pack2(p, p + 8);
}

// B-matrix 32x16 fragment where source stores B^T row-major: row n holds B[k][n],
// k contiguous. Contiguous k layout (as load_afrag): lanes 0-15 K=0..15, lanes 16-31 K=16..31, N=lane%16.
__device__ __forceinline__ half16 load_bfrag(const _Float16* base, int ld, int lane) {
  const int n  = lane & 15;
  const int kg = (lane & 16) ? 16 : 0;
  const _Float16* p = base + (size_t)n * ld + kg;
  return pack2(p, p + 8);
}

// ---------------- fp32 -> f16 cast of x, qkv_w, out_w ----------------
__global__ void __launch_bounds__(256)
cvt_kernel(const float* __restrict__ x, const float* __restrict__ qw,
           const float* __restrict__ ow, _Float16* __restrict__ xh,
           _Float16* __restrict__ qwh, _Float16* __restrict__ owh) {
  const int n1 = M_ * D_, n2 = N1_ * D_, n3 = D_ * D_;
  const int total = n1 + n2 + n3;
  int idx = blockIdx.x * blockDim.x + threadIdx.x;
  if (idx >= total) return;
  if (idx < n1)           xh[idx] = (_Float16)x[idx];
  else if (idx < n1 + n2) qwh[idx - n1] = (_Float16)qw[idx - n1];
  else                    owh[idx - n1 - n2] = (_Float16)ow[idx - n1 - n2];
}

// ------------- GEMM: C[M,N] = A[M,K] * W[N,K]^T + bias -------------
// One wave computes a 32x32 C tile: 4 WMMAs per 32-wide K step.
__global__ void __launch_bounds__(256)
gemm_wmma(const _Float16* __restrict__ A, const _Float16* __restrict__ W,
          const float* __restrict__ bias, _Float16* __restrict__ Ch,
          float* __restrict__ Cf, int M, int N, int K, int store_f32) {
  const int lane = threadIdx.x & 31;
  const int wid  = blockIdx.x * (blockDim.x >> 5) + (threadIdx.x >> 5);
  const int Mt = M >> 5, Nt = N >> 5;
  if (wid >= Mt * Nt) return;
  const int m0 = (wid % Mt) << 5;
  const int n0 = (wid / Mt) << 5;

  float8 acc00 = {}, acc01 = {}, acc10 = {}, acc11 = {};
  for (int k0 = 0; k0 < K; k0 += 32) {
    half16 a0 = load_afrag(A + (size_t)m0 * K + k0, K, lane);
    half16 a1 = load_afrag(A + (size_t)(m0 + 16) * K + k0, K, lane);
    half16 b0 = load_bfrag(W + (size_t)n0 * K + k0, K, lane);
    half16 b1 = load_bfrag(W + (size_t)(n0 + 16) * K + k0, K, lane);
    acc00 = wmma16(a0, b0, acc00);
    acc01 = wmma16(a0, b1, acc01);
    acc10 = wmma16(a1, b0, acc10);
    acc11 = wmma16(a1, b1, acc11);
  }

  float8 accs[2][2] = {{acc00, acc01}, {acc10, acc11}};
  const int nlo = lane & 15;
  const int rsh = (lane & 16) ? 8 : 0;
#pragma unroll
  for (int im = 0; im < 2; ++im)
#pragma unroll
    for (int in = 0; in < 2; ++in) {
      const int nn = n0 + in * 16 + nlo;
      const float bv = bias[nn];
#pragma unroll
      for (int r = 0; r < 8; ++r) {
        const int mm = m0 + im * 16 + r + rsh;
        const float v = accs[im][in][r] + bv;
        if (store_f32) Cf[(size_t)mm * N + nn] = v;
        else           Ch[(size_t)mm * N + nn] = (_Float16)v;
      }
    }
}

// ------------- strided-sparse flash attention -------------
// One wave per (b, h, 16-query tile). Key blocks of 32:
//   compact blocks: key c -> j = 8*(c0+c), mask j <= i       (all strided keys)
//   local block:    key c -> j = jbase+c,  mask (j==i||j==i-1)&&(j%8!=0)
__global__ void __launch_bounds__(128)
attn_kernel(const _Float16* __restrict__ qkvh, _Float16* __restrict__ aoh) {
  __shared__ __align__(16) _Float16 Pls[4][16 * 32];  // probs   16q x 32k
  __shared__ __align__(16) _Float16 Vls[4][64 * 32];  // V^T     64d x 32k
  const int lane = threadIdx.x & 31;
  const int wv   = threadIdx.x >> 5;
  const int gw   = blockIdx.x * 4 + wv;
  const int qt = gw & 127;            // 128 query tiles
  const int hh = (gw >> 7) % H_;
  const int bb = gw / (128 * H_);
  const int i0 = qt * 16;
  _Float16* P  = Pls[wv];
  _Float16* Vt = Vls[wv];
  const size_t rowb = (size_t)(bb * S_) * (size_t)N1_;
  const float scale = 0.125f;  // 64^-0.5

  const _Float16* qbase = qkvh + rowb + (size_t)i0 * N1_ + hh * HD_;
  half16 qf0 = load_afrag(qbase,      N1_, lane);  // d 0..31
  half16 qf1 = load_afrag(qbase + 32, N1_, lane);  // d 32..63

  float8 O[4] = {};
  float mrow[8], lrow[8];
#pragma unroll
  for (int r = 0; r < 8; ++r) { mrow[r] = -INFINITY; lrow[r] = 0.f; }

  const int colc   = lane & 15;
  const int rowoff = (lane & 16) ? 8 : 0;

  const int Cmax     = (i0 + 15) >> 3;      // largest strided index needed
  const int nCompact = (Cmax >> 5) + 1;
  const int jlocal   = (i0 >= 16) ? (i0 - 16) : 0;

  for (int blk = 0; blk <= nCompact; ++blk) {
    const bool compact = (blk < nCompact);
    const int j00   = compact ? (blk << 8) : jlocal;  // blk*32*8
    const int jstep = compact ? 8 : 1;

    // ---- stage V tile transposed into LDS (lane = key column) ----
    {
      const int c = lane;
      const int j = j00 + jstep * c;
      const _Float16* vp = qkvh + rowb + (size_t)j * N1_ + 2 * D_ + hh * HD_;
#pragma unroll
      for (int ch = 0; ch < 8; ++ch) {
        half8 vv = *(const half8*)(vp + ch * 8);
#pragma unroll
        for (int d = 0; d < 8; ++d) Vt[(ch * 8 + d) * 32 + c] = vv[d];
      }
    }

    // ---- scores: two 16x16 tiles covering 32 keys ----
    const int jA = j00 + jstep * colc;
    const int jB = j00 + jstep * (colc + 16);
    float8 s0 = {}, s1 = {};
#pragma unroll
    for (int dstep = 0; dstep < 2; ++dstep) {
      const int kcol = D_ + hh * HD_ + dstep * 32 + ((lane & 16) ? 16 : 0);
      const _Float16* pa = qkvh + rowb + (size_t)jA * N1_ + kcol;
      const _Float16* pb = qkvh + rowb + (size_t)jB * N1_ + kcol;
      half16 kfa = pack2(pa, pa + 8);
      half16 kfb = pack2(pb, pb + 8);
      s0 = wmma16(dstep ? qf1 : qf0, kfa, s0);
      s1 = wmma16(dstep ? qf1 : qf0, kfb, s1);
    }

    // ---- mask + scale ----
    float sm0[8], sm1[8], red[8];
#pragma unroll
    for (int r = 0; r < 8; ++r) {
      const int i = i0 + r + rowoff;
      bool a0, a1;
      if (compact) { a0 = (jA <= i); a1 = (jB <= i); }
      else {
        a0 = ((jA == i) || (jA == i - 1)) && ((jA & 7) != 0);
        a1 = ((jB == i) || (jB == i - 1)) && ((jB & 7) != 0);
      }
      sm0[r] = a0 ? s0[r] * scale : -INFINITY;
      sm1[r] = a1 ? s1[r] * scale : -INFINITY;
      red[r] = fmaxf(sm0[r], sm1[r]);
    }
    // row max across the 16 lanes of each half (rows stay within a half)
#pragma unroll
    for (int off = 8; off > 0; off >>= 1)
#pragma unroll
      for (int r = 0; r < 8; ++r)
        red[r] = fmaxf(red[r], __shfl_xor(red[r], off, 32));

    float alpha[8];
#pragma unroll
    for (int r = 0; r < 8; ++r) {
      const float mn = fmaxf(mrow[r], red[r]);
      alpha[r] = __expf(mrow[r] - mn);   // first block: exp(-inf)=0, no NaN
      mrow[r] = mn;
      lrow[r] *= alpha[r];
    }

    // ---- probabilities -> LDS (f16) + row sums ----
#pragma unroll
    for (int r = 0; r < 8; ++r) {
      const float p0 = __expf(sm0[r] - mrow[r]);
      const float p1 = __expf(sm1[r] - mrow[r]);
      P[(r + rowoff) * 32 + colc]      = (_Float16)p0;
      P[(r + rowoff) * 32 + colc + 16] = (_Float16)p1;
      red[r] = p0 + p1;
    }
#pragma unroll
    for (int off = 8; off > 0; off >>= 1)
#pragma unroll
      for (int r = 0; r < 8; ++r)
        red[r] += __shfl_xor(red[r], off, 32);
#pragma unroll
    for (int r = 0; r < 8; ++r) lrow[r] += red[r];

    asm volatile("s_wait_dscnt 0x0" ::: "memory");  // LDS stores visible to reads

    // ---- O = alpha*O + P @ V ----
    half16 pf = load_afrag(P, 32, lane);
#pragma unroll
    for (int db = 0; db < 4; ++db) {
      half16 vf = load_bfrag(Vt + db * 16 * 32, 32, lane);
#pragma unroll
      for (int r = 0; r < 8; ++r) O[db][r] *= alpha[r];
      O[db] = wmma16(pf, vf, O[db]);
    }
    asm volatile("" ::: "memory");  // keep LDS reuse ordered across iterations
  }

  // ---- finalize: divide by softmax sum, store f16 [B*S, 768] ----
  _Float16* obase = aoh + (size_t)(bb * S_ + i0) * D_ + hh * HD_;
#pragma unroll
  for (int r = 0; r < 8; ++r) {
    const float inv = 1.0f / lrow[r];   // diagonal always present => lrow > 0
    const int i = r + rowoff;
#pragma unroll
    for (int db = 0; db < 4; ++db)
      obase[(size_t)i * D_ + db * 16 + colc] = (_Float16)(O[db][r] * inv);
  }
}

extern "C" void kernel_launch(void* const* d_in, const int* in_sizes, int n_in,
                              void* d_out, int out_size, void* d_ws, size_t ws_size,
                              hipStream_t stream) {
  (void)in_sizes; (void)n_in; (void)out_size; (void)ws_size;
  const float* x  = (const float*)d_in[0];
  const float* qw = (const float*)d_in[1];
  const float* qb = (const float*)d_in[2];
  const float* ow = (const float*)d_in[3];
  const float* ob = (const float*)d_in[4];
  float* out = (float*)d_out;

  char* ws = (char*)d_ws;
  _Float16* xh   = (_Float16*)ws; ws += (size_t)M_ * D_ * 2;
  _Float16* qwh  = (_Float16*)ws; ws += (size_t)N1_ * D_ * 2;
  _Float16* owh  = (_Float16*)ws; ws += (size_t)D_ * D_ * 2;
  _Float16* qkvh = (_Float16*)ws; ws += (size_t)M_ * N1_ * 2;
  _Float16* aoh  = (_Float16*)ws;

  {
    const int total = M_ * D_ + N1_ * D_ + D_ * D_;
    cvt_kernel<<<(total + 255) / 256, 256, 0, stream>>>(x, qw, ow, xh, qwh, owh);
  }
  {  // QKV projection: [4096,2304] = xh * qkv_w^T + qkv_b  -> f16
    const int waves = (M_ / 32) * (N1_ / 32);
    gemm_wmma<<<(waves + 7) / 8, 256, 0, stream>>>(xh, qwh, qb, qkvh, nullptr,
                                                   M_, N1_, D_, 0);
  }
  {  // strided-sparse attention -> aoh f16 [4096,768]
    const int waves = B_ * H_ * (S_ / 16);
    attn_kernel<<<waves / 4, 128, 0, stream>>>(qkvh, aoh);
  }
  {  // output projection: d_out = aoh * out_w^T + out_b  -> f32
    const int waves = (M_ / 32) * (D_ / 32);
    gemm_wmma<<<(waves + 7) / 8, 256, 0, stream>>>(aoh, owh, ob, nullptr, out,
                                                   M_, D_, D_, 1);
  }
}
